// WordByWordSlideSumInsideSentence_23347442221549
// MI455X (gfx1250) — matrix operand order repaired and checked by reading, not compiled, not run on hardware
//
#include <hip/hip_runtime.h>
#include <hip/hip_bf16.h>
#include <math.h>

typedef _Float16 half_t;
typedef __attribute__((ext_vector_type(16))) _Float16 v16h;
typedef __attribute__((ext_vector_type(8)))  _Float16 v8h;
typedef __attribute__((ext_vector_type(8)))  float    v8f;

#define B_BATCH 128
#define W1      2048
#define DIM     300
#define KP      320      // padded K (10 x 32)
#define NA      128      // answer words per batch (4 x 32)
#define WINDOW  60
#define STRIDE  3
#define NWIN    663      // (2048-60)/3 + 1
#define JPB     64       // windows per block
#define NJB     11       // ceil(663/64)
#define SROWS   256      // story rows per block (covers 3*63+60 = 249)
#define SA      328      // LDS A stride in halfs (pad to break conflicts)
#define SC      132      // LDS cos stride in floats (pad to break conflicts)

#define LDS_A_BYTES   (SROWS * SA * 2)        // 167936
#define LDS_C_BYTES   (SROWS * SC * 4)        // 135168
#define LDS_W_BYTES   (64 * 4)
#define SHMEM_BYTES   (LDS_A_BYTES + LDS_C_BYTES + LDS_W_BYTES)  // 303360 <= 320KB

// ---------------------------------------------------------------------------
// Kernel 1: L2-normalize answer vectors -> f16 B matrix [b][a][k], K padded 320
// one wave per row
// ---------------------------------------------------------------------------
__global__ __launch_bounds__(256) void norm_answers_kernel(
    const float* __restrict__ la, half_t* __restrict__ Bmat, int nrows) {
  int gwave = (int)((blockIdx.x * blockDim.x + threadIdx.x) >> 5);
  int lane  = (int)(threadIdx.x & 31);
  if (gwave >= nrows) return;
  const float* row = la + (size_t)gwave * DIM;
  float vals[10];
  float ss = 0.f;
#pragma unroll
  for (int q = 0; q < 10; ++q) {
    int d = lane + 32 * q;
    float v = (d < DIM) ? row[d] : 0.f;
    vals[q] = v;
    ss += v * v;
  }
#pragma unroll
  for (int off = 16; off; off >>= 1) ss += __shfl_xor(ss, off, 32);
  float rn = rsqrtf(ss + 1e-6f);
  half_t* out = Bmat + (size_t)gwave * KP;
#pragma unroll
  for (int q = 0; q < 10; ++q) {
    int d = lane + 32 * q;                 // 0..319, zero-padded past 300
    out[d] = (half_t)(vals[q] * rn);
  }
}

// ---------------------------------------------------------------------------
// Kernel 2: fused normalize-story + WMMA cos GEMM + gaussian window max/mean
// grid.x = batch * NJB + window-chunk ; 256 threads (8 waves)
// ---------------------------------------------------------------------------
__global__ __launch_bounds__(256) void wbw_slide_kernel(
    const float* __restrict__ ls, const half_t* __restrict__ Bmat,
    const float* __restrict__ wg, float* __restrict__ part) {
  extern __shared__ char smem[];
  half_t* ldsA = (half_t*)smem;
  float*  ldsC = (float*)(smem + LDS_A_BYTES);
  float*  wl   = (float*)(smem + LDS_A_BYTES + LDS_C_BYTES);
  float*  red  = (float*)smem;   // aliases ldsA after GEMM is done

  const int b     = (int)blockIdx.x / NJB;
  const int jb    = (int)blockIdx.x % NJB;
  const int t     = (int)threadIdx.x;
  const int lane  = t & 31;
  const int wave  = t >> 5;
  const int sbase = jb * (JPB * STRIDE);   // first story word of this chunk

  if (t < WINDOW) wl[t] = wg[t];

  // ---- phase 1: load + L2-normalize 256 story rows into LDS as f16 ----
  const float* lsb = ls + (size_t)b * W1 * DIM;
  for (int i = 0; i < 32; ++i) {
    int row = i * 8 + wave;
    int s   = sbase + row;
    bool ok = (s < W1);
    const float* rp = lsb + (size_t)(ok ? s : 0) * DIM;
    float vals[10];
    float ss = 0.f;
#pragma unroll
    for (int q = 0; q < 10; ++q) {
      int d = lane + 32 * q;
      float v = (ok && d < DIM) ? rp[d] : 0.f;
      vals[q] = v;
      ss += v * v;
    }
#pragma unroll
    for (int off = 16; off; off >>= 1) ss += __shfl_xor(ss, off, 32);
    float rn = rsqrtf(ss + 1e-6f);
    half_t* ar = ldsA + row * SA;
#pragma unroll
    for (int q = 0; q < 10; ++q) ar[lane + 32 * q] = (half_t)(vals[q] * rn);
  }
  __syncthreads();

  // ---- phase 2: 256x128x320 f16 WMMA GEMM, cos tile -> ldsC ----
  {
    const int mt0     = wave * 2;          // 2 M-tiles per wave, 8 N-tiles
    const int r16     = lane & 15;
    const int halfsel = lane >> 4;         // 0: low half of lanes, 1: high
    const half_t* Bb  = Bmat + (size_t)b * NA * KP;

    v8f acc[2][8] = {};

    for (int kb = 0; kb < 10; ++kb) {
      // A fragment: lane holds row M=r16 (+tile), K = k0..k0+15 contiguous
      int k0 = kb * 32 + halfsel * 16;
      v16h afr[2];
#pragma unroll
      for (int mi = 0; mi < 2; ++mi) {
        const half_t* ap = ldsA + ((mt0 + mi) * 16 + r16) * SA + k0;
        v8h lo = *(const v8h*)(ap);
        v8h hi = *(const v8h*)(ap + 8);
        afr[mi] = __builtin_shufflevector(lo, hi, 0, 1, 2, 3, 4, 5, 6, 7,
                                          8, 9, 10, 11, 12, 13, 14, 15);
      }
      // B fragment: lane holds column N=r16 (+tile), K contiguous 16
      int kb0 = kb * 32 + halfsel * 16;
#pragma unroll
      for (int nt = 0; nt < 8; ++nt) {
        const half_t* bp = Bb + (size_t)(nt * 16 + r16) * KP + kb0;
        v16h bfr = *(const v16h*)(bp);
        acc[0][nt] = __builtin_amdgcn_wmma_f32_16x16x32_f16(
            false, afr[0], false, bfr, (short)0, acc[0][nt], false, false);
        acc[1][nt] = __builtin_amdgcn_wmma_f32_16x16x32_f16(
            false, afr[1], false, bfr, (short)0, acc[1][nt], false, false);
      }
    }

    // C/D layout: VGPR r -> M = r + 8*halfsel, N = lane%16
#pragma unroll
    for (int mi = 0; mi < 2; ++mi) {
      int mrow0 = (mt0 + mi) * 16 + halfsel * 8;
#pragma unroll
      for (int nt = 0; nt < 8; ++nt) {
        int col = nt * 16 + r16;
#pragma unroll
        for (int r = 0; r < 8; ++r) ldsC[(mrow0 + r) * SC + col] = acc[mi][nt][r];
      }
    }
  }
  __syncthreads();

  // ---- phase 3: gaussian-weighted sliding max + masked mean + chunk max ----
  {
    int jl = t >> 2;           // local window 0..63
    int m  = t & 3;            // answer sentence (N2)
    int j  = jb * JPB + jl;
    float avg = -INFINITY;
    if (j < NWIN) {
      float sum = 0.f, cnt = 0.f;
      for (int v = 0; v < 32; ++v) {
        const float* cp = ldsC + (STRIDE * jl) * SC + m * 32 + v;
        float wm = -INFINITY;
#pragma unroll
        for (int k = 0; k < WINDOW; ++k) wm = fmaxf(wm, wl[k] * cp[k * SC]);
        sum += wm;
        cnt += (wm != 0.f) ? 1.f : 0.f;
      }
      avg = sum / (cnt + 1e-5f);
    }
    __syncthreads();           // done reading ldsA region; safe to alias as red
    red[t] = avg;
    __syncthreads();
    if (t < 4) {
      float mx = -INFINITY;
      for (int q = 0; q < JPB; ++q) mx = fmaxf(mx, red[q * 4 + t]);
      part[((size_t)b * NJB + jb) * 4 + t] = mx;
    }
  }
}

// ---------------------------------------------------------------------------
// Kernel 3: reduce chunk partials, scale by alpha -> out (128,1,4)
// ---------------------------------------------------------------------------
__global__ __launch_bounds__(256) void finalize_kernel(
    const float* __restrict__ part, const float* __restrict__ alpha,
    float* __restrict__ out) {
  int idx = (int)(blockIdx.x * blockDim.x + threadIdx.x);
  if (idx >= B_BATCH * 4) return;
  int b = idx >> 2, m = idx & 3;
  float mx = -INFINITY;
  for (int jb = 0; jb < NJB; ++jb)
    mx = fmaxf(mx, part[((size_t)b * NJB + jb) * 4 + m]);
  out[idx] = mx * alpha[0];
}

extern "C" void kernel_launch(void* const* d_in, const int* in_sizes, int n_in,
                              void* d_out, int out_size, void* d_ws, size_t ws_size,
                              hipStream_t stream) {
  const float* ls    = (const float*)d_in[0];   // (128,1,2048,300)
  const float* la    = (const float*)d_in[1];   // (128,4,32,300)
  const float* wg    = (const float*)d_in[2];   // (60,)
  const float* alpha = (const float*)d_in[3];   // scalar
  float* out = (float*)d_out;                   // 512 floats

  // workspace: [ f16 B matrix 128*128*320 | f32 partials 128*11*4 ]
  half_t* Bmat = (half_t*)d_ws;
  float*  part = (float*)((char*)d_ws + (size_t)B_BATCH * NA * KP * sizeof(half_t));

  // 1) normalize answers: 16384 rows, one wave each -> 2048 blocks of 8 waves
  norm_answers_kernel<<<2048, 256, 0, stream>>>(la, Bmat, B_BATCH * NA);

  // 2) fused cos-GEMM + windowing: 128 batches x 11 window-chunks
  wbw_slide_kernel<<<B_BATCH * NJB, 256, SHMEM_BYTES, stream>>>(ls, Bmat, wg, part);

  // 3) final reduce + alpha
  finalize_kernel<<<2, 256, 0, stream>>>(part, alpha, out);
}
